// BindRegressionModel_84636625535189
// MI455X (gfx1250) — compile-verified, matrix operand order checked by reading, not run on hardware
//
#include <hip/hip_runtime.h>
#include <hip/hip_bf16.h>

#define B_   64
#define M_   128
#define NN_  8192        // B*M nodes
#define E_   32768
#define ET_  (E_ + NN_)  // edges + self loops
#define L_   384
#define PD_  1280
#define D_   64
#define H_   16
#define NEG_ 0.05f
#define EPS_ 1e-5f

typedef __attribute__((ext_vector_type(16))) _Float16 v16h;
typedef __attribute__((ext_vector_type(8)))  _Float16 v8h;
typedef __attribute__((ext_vector_type(4)))  _Float16 h4;
typedef __attribute__((ext_vector_type(8)))  float    v8f;

__device__ __forceinline__ float leaky_(float v){ return v >= 0.f ? v : NEG_ * v; }
__device__ __forceinline__ int   fordkey(float f){ int i = __float_as_int(f); return i >= 0 ? i : i ^ 0x7fffffff; }
__device__ __forceinline__ float fordinv(int i){ return __int_as_float(i >= 0 ? i : i ^ 0x7fffffff); }

// ---------------------------------------------------------------------------
// WMMA GEMM:  C = epi(A[M x K](f32) * Wt[N x K](f16,pretransposed) + bias)
// Requirements (all launches satisfy them): Mrows % 64 == 0, K % 32 == 0,
// N an exact multiple of NT*16. Block = 128 threads = 4 waves; block tile
// 64 x (NT*16); wave w owns rows [16w,16w+16).
// Double-buffered LDS, last iteration peeled: one barrier per k-step and no
// conditional staging. Staging issues ALL global loads before any convert/
// ds_store so loadcnt waits amortize across 4+NT/2 outstanding loads.
// AMODE_COMBINE: K=2048 row-gather (first 1024 cols from hd[row], second
// from hd[row^1]); EPI_TANHPAIR averages tanh of adjacent row pairs and
// writes Mrows/2 output rows (the symmetric combine step, fused).
// ---------------------------------------------------------------------------
enum { A_PLAIN = 0, A_COMBINE = 1 };
enum { EPI_BIAS = 0, EPI_TANHPAIR = 1, EPI_RESID = 2 };

template<int AMODE, int EPI, int NT>
__launch_bounds__(128)
__global__ void gemm_wmma(const float* __restrict__ A, const _Float16* __restrict__ Wt,
                          const float* __restrict__ bias, const float* __restrict__ resid,
                          float* __restrict__ C, int Mrows, int K, int N)
{
    __shared__ __align__(16) _Float16 sA[2][64 * 32];
    __shared__ __align__(16) _Float16 sB[2][NT * 16 * 32];

    const int tid  = threadIdx.x;
    const int lane = tid & 31;
    const int wave = tid >> 5;
    const int rowBase = blockIdx.x * 64;
    const int colBase = blockIdx.y * (NT * 16);

    v8f acc[NT];
#pragma unroll
    for (int i = 0; i < NT; i++)
#pragma unroll
        for (int j = 0; j < 8; j++) acc[i][j] = 0.f;

    // stage one 32-wide k-slab into buffer p: load phase, then store phase
    auto stage = [&](int k0, int p) {
        float4 va[4];
#pragma unroll
        for (int it = 0; it < 4; it++) {          // A: 512 chunks of 4 floats
            int c  = tid + it * 128;
            int r  = c >> 3, kk = (c & 7) << 2;
            const float* ap;
            if (AMODE == A_PLAIN) {
                ap = A + (size_t)(rowBase + r) * K + (k0 + kk);
            } else {
                int row = rowBase + r;
                int kc  = k0 + kk;
                int src = (kc < 1024) ? row : (row ^ 1);
                if (kc >= 1024) kc -= 1024;
                ap = A + (size_t)src * 1024 + kc;
            }
            // speculative prefetch 2 slabs ahead (OOB prefetch dropped by HW)
            __builtin_prefetch((const char*)ap + 256, 0, 1);
            va[it] = *(const float4*)ap;
        }
        v8h vbv[NT / 2];
#pragma unroll
        for (int it = 0; it < NT / 2; it++) {     // B: NT*64 chunks of 8 halves
            int c   = tid + it * 128;
            int col = c >> 2, kk = (c & 3) << 3;
            vbv[it] = *(const v8h*)(Wt + (size_t)(colBase + col) * K + (k0 + kk));
        }
#pragma unroll
        for (int it = 0; it < 4; it++) {
            int c  = tid + it * 128;
            int r  = c >> 3, kk = (c & 7) << 2;
            h4 hv; hv[0] = (_Float16)va[it].x; hv[1] = (_Float16)va[it].y;
                   hv[2] = (_Float16)va[it].z; hv[3] = (_Float16)va[it].w;
            *(h4*)&sA[p][r * 32 + kk] = hv;
        }
#pragma unroll
        for (int it = 0; it < NT / 2; it++) {
            int c   = tid + it * 128;
            int col = c >> 2, kk = (c & 3) << 3;
            *(v8h*)&sB[p][col * 32 + kk] = vbv[it];
        }
    };

    const int rsel = lane & 15;
    const int kb   = (lane >> 4) << 3;
    auto computeStep = [&](int p) {
        // fragments per documented 16-bit 16x32 layout
        v8h alo = *(const v8h*)&sA[p][(wave * 16 + rsel) * 32 + kb];
        v8h ahi = *(const v8h*)&sA[p][(wave * 16 + rsel) * 32 + kb + 16];
        v16h afrag;
#pragma unroll
        for (int i = 0; i < 8; i++) { afrag[i] = alo[i]; afrag[i + 8] = ahi[i]; }
#pragma unroll
        for (int nt = 0; nt < NT; nt++) {
            v8h blo = *(const v8h*)&sB[p][(nt * 16 + rsel) * 32 + kb];
            v8h bhi = *(const v8h*)&sB[p][(nt * 16 + rsel) * 32 + kb + 16];
            v16h bfrag;
#pragma unroll
            for (int i = 0; i < 8; i++) { bfrag[i] = blo[i]; bfrag[i + 8] = bhi[i]; }
            acc[nt] = __builtin_amdgcn_wmma_f32_16x16x32_f16(
                false, afrag, false, bfrag, (short)0, acc[nt], false, false);
        }
    };

    const int nsteps = K >> 5;
    stage(0, 0);
    __syncthreads();
    int p = 0;
    for (int s = 0; s + 1 < nsteps; s++) {        // steady state: stage-ahead
        stage((s + 1) << 5, p ^ 1);
        computeStep(p);
        __syncthreads();
        p ^= 1;
    }
    computeStep(p);                               // peeled last slab

    // ---- epilogue: lane<16 -> (M=g, N=lane); lane>=16 -> (M=g+8, N=lane-16)
    const int nloc = lane & 15;
    const int mhi  = (lane >> 4) << 3;
#pragma unroll
    for (int nt = 0; nt < NT; nt++) {
        int col = colBase + nt * 16 + nloc;
        float bv = bias[col];
        if (EPI == EPI_TANHPAIR) {
            int base = rowBase + wave * 16 + mhi;       // even
#pragma unroll
            for (int j = 0; j < 4; j++) {
                float va = tanhf(acc[nt][2 * j]     + bv);
                float vb = tanhf(acc[nt][2 * j + 1] + bv);
                C[(size_t)(base / 2 + j) * N + col] = 0.5f * (va + vb);
            }
        } else {
#pragma unroll
            for (int g = 0; g < 8; g++) {
                int row = rowBase + wave * 16 + mhi + g;
                float v = acc[nt][g] + bv;
                if (EPI == EPI_RESID) v += resid[(size_t)row * N + col];
                C[(size_t)row * N + col] = v;
            }
        }
    }
}

// ---------------------------------------------------------------------------
__global__ void fill_u32(unsigned* p, unsigned v, int n)
{
    int t = blockIdx.x * 256 + threadIdx.x;
    if (t < n) p[t] = v;
}

// transpose + convert weight: W[K][N] f32 -> Wt[N][Kpad] f16 (zero pad K..Kpad)
__global__ void wt_prep(const float* __restrict__ W, _Float16* __restrict__ Wt,
                        int K, int N, int Kpad)
{
    int t = blockIdx.x * 256 + threadIdx.x;
    if (t >= N * Kpad) return;
    int n = t / Kpad, k = t - n * Kpad;
    Wt[t] = (k < K) ? (_Float16)W[(size_t)k * N + n] : (_Float16)0.f;
}

// fused K|V weight: t_wkv[n][k], n<128, k<1280 ; plus bkv[128]
__global__ void kv_wt_prep(const float* __restrict__ wk, const float* __restrict__ wv,
                           const float* __restrict__ bk, const float* __restrict__ bv,
                           _Float16* __restrict__ wkv, float* __restrict__ bkv)
{
    int t = blockIdx.x * 256 + threadIdx.x;
    if (t < 128 * PD_) {
        int n = t / PD_, k = t - n * PD_;
        wkv[t] = (_Float16)((n < 64) ? wk[(size_t)k * 64 + n] : wv[(size_t)k * 64 + (n - 64)]);
    }
    if (t < 128) bkv[t] = (t < 64) ? bk[t] : bv[t - 64];
}

// node embed -> h16 stored 8192 x 32 (upper 16 cols zero for K=32 WMMA)
__global__ void node_embed(const float* __restrict__ x, const float* __restrict__ w,
                           const float* __restrict__ bc, const float* __restrict__ g,
                           const float* __restrict__ b2, float* __restrict__ h16)
{
    int b = blockIdx.x, tid = threadIdx.x;
    __shared__ float buf[2048];
    __shared__ float r1[256], r2[256];
    float ls = 0.f, lq = 0.f;
    for (int idx = tid; idx < 2048; idx += 256) {
        int n = idx >> 4, f = idx & 15;
        const float* xp = x + (size_t)(b * 128 + n) * 15;
        float acc = bc[f];
        for (int k = 0; k < 15; k++) acc += xp[k] * w[k * 16 + f];
        acc = leaky_(acc);
        buf[idx] = acc; ls += acc; lq += acc * acc;
    }
    r1[tid] = ls; r2[tid] = lq; __syncthreads();
    for (int s = 128; s > 0; s >>= 1) {
        if (tid < s) { r1[tid] += r1[tid + s]; r2[tid] += r2[tid + s]; }
        __syncthreads();
    }
    float mu = r1[0] / 2048.f, var = r2[0] / 2048.f - mu * mu;
    float rs = rsqrtf(var + EPS_);
    for (int idx = tid; idx < 2048; idx += 256) {
        int n = idx >> 4, f = idx & 15;
        h16[(size_t)b * 4096 + n * 32 + f]      = (buf[idx] - mu) * rs * g[f] + b2[f];
        h16[(size_t)b * 4096 + n * 32 + 16 + f] = 0.f;
    }
}

__global__ void edge_embed(const float* __restrict__ efi, const float* __restrict__ w,
                           const float* __restrict__ bb, const float* __restrict__ g,
                           const float* __restrict__ b2, float* __restrict__ ef)
{
    int e = blockIdx.x * 256 + threadIdx.x;
    if (e >= E_) return;
    const float* p = efi + (size_t)e * 5;
    float v0 = bb[0], v1 = bb[1];
    for (int k = 0; k < 5; k++) { v0 += p[k] * w[k * 2]; v1 += p[k] * w[k * 2 + 1]; }
    v0 = leaky_(v0); v1 = leaky_(v1);
    float mu = 0.5f * (v0 + v1);
    float var = 0.5f * ((v0 - mu) * (v0 - mu) + (v1 - mu) * (v1 - mu));
    float rs = rsqrtf(var + EPS_);
    ef[e * 2]     = (v0 - mu) * rs * g[0] + b2[0];
    ef[e * 2 + 1] = (v1 - mu) * rs * g[1] + b2[1];
}

__global__ void gat_prep(const int* __restrict__ dst, const float* __restrict__ ef,
                         float* __restrict__ cnt, float* __restrict__ la)
{
    int e = blockIdx.x * 256 + threadIdx.x;
    if (e >= E_) return;
    int d = dst[e];
    atomicAdd(&cnt[d], 1.f);
    atomicAdd(&la[d * 2],     ef[e * 2]);
    atomicAdd(&la[d * 2 + 1], ef[e * 2 + 1]);
}

__global__ void gat_logits(const int* __restrict__ srcp, const int* __restrict__ dstp,
                           const float* __restrict__ ef, const float* __restrict__ cnt,
                           const float* __restrict__ la, const float* __restrict__ xl,
                           const float* __restrict__ xr, const float* __restrict__ we,
                           const float* __restrict__ att, float* __restrict__ logit,
                           int* __restrict__ s2, int* __restrict__ d2, int* __restrict__ mxi)
{
    int e = blockIdx.x * 256 + threadIdx.x;
    if (e >= ET_) return;
    int s, d; float a0, a1;
    if (e < E_) { s = srcp[e]; d = dstp[e]; a0 = ef[e * 2]; a1 = ef[e * 2 + 1]; }
    else {
        int n = e - E_; s = n; d = n;
        float c = fmaxf(cnt[n], 1.f);
        a0 = la[n * 2] / c; a1 = la[n * 2 + 1] / c;
    }
    const float* pl = xl + (size_t)s * 64;
    const float* pr = xr + (size_t)d * 64;
    float lg = 0.f;
    for (int f = 0; f < 64; f++) {
        float v = pl[f] + pr[f] + a0 * we[f] + a1 * we[64 + f];
        lg += leaky_(v) * att[f];
    }
    logit[e] = lg; s2[e] = s; d2[e] = d;
    atomicMax(&mxi[d], fordkey(lg));
}

__global__ void gat_exp(const float* __restrict__ logit, const int* __restrict__ d2,
                        const int* __restrict__ mxi, float* __restrict__ av,
                        float* __restrict__ den)
{
    int e = blockIdx.x * 256 + threadIdx.x;
    if (e >= ET_) return;
    float a = __expf(logit[e] - fordinv(mxi[d2[e]]));
    av[e] = a;
    atomicAdd(&den[d2[e]], a);
}

__global__ void gat_scatter(const float* __restrict__ av, const float* __restrict__ den,
                            const int* __restrict__ s2, const int* __restrict__ d2,
                            const float* __restrict__ xl, float* __restrict__ hout)
{
    int t = blockIdx.x * 256 + threadIdx.x;
    int e = t >> 6, f = t & 63;
    if (e >= ET_) return;
    float alpha = av[e] / den[d2[e]];
    atomicAdd(&hout[(size_t)d2[e] * 64 + f], alpha * xl[(size_t)s2[e] * 64 + f]);
}

__global__ void gat_ln(const float* __restrict__ hout, const float* __restrict__ bias,
                       const float* __restrict__ g, const float* __restrict__ b2,
                       float* __restrict__ h)
{
    int b = blockIdx.x, tid = threadIdx.x;
    __shared__ float r1[256], r2[256];
    float ls = 0.f, lq = 0.f;
    for (int idx = tid; idx < 8192; idx += 256) {
        float v = leaky_(hout[(size_t)b * 8192 + idx] + bias[idx & 63]);
        ls += v; lq += v * v;
    }
    r1[tid] = ls; r2[tid] = lq; __syncthreads();
    for (int s = 128; s > 0; s >>= 1) {
        if (tid < s) { r1[tid] += r1[tid + s]; r2[tid] += r2[tid + s]; }
        __syncthreads();
    }
    float mu = r1[0] / 8192.f, var = r2[0] / 8192.f - mu * mu;
    float rs = rsqrtf(var + EPS_);
    for (int idx = tid; idx < 8192; idx += 256) {
        float v = leaky_(hout[(size_t)b * 8192 + idx] + bias[idx & 63]);
        h[(size_t)b * 8192 + idx] = (v - mu) * rs * g[idx & 63] + b2[idx & 63];
    }
}

// online-softmax cross attention (DH=4/head), vectorized KV reads
__global__ void attn_softmax(const float* __restrict__ q, const float* __restrict__ kv,
                             const unsigned char* __restrict__ mask, float* __restrict__ out)
{
    int t = blockIdx.x * 256 + threadIdx.x;   // B*M*H threads
    if (t >= NN_ * H_) return;
    int hh = t & 15, nm = t >> 4, b = nm >> 7;  // nm: node, b: batch (128 nodes/batch)
    const float4 qv = *(const float4*)(q + (size_t)nm * 64 + hh * 4);
    float q0 = qv.x * 0.5f, q1 = qv.y * 0.5f, q2 = qv.z * 0.5f, q3 = qv.w * 0.5f;
    const float4* kvb = (const float4*)(kv + (size_t)b * L_ * 128) + hh;  // 32 float4 per row
    const unsigned char* mb = mask + (size_t)b * L_;
    float mx = -1e30f, den = 0.f, a0 = 0.f, a1 = 0.f, a2 = 0.f, a3 = 0.f;
    for (int l = 0; l < L_; l++) {
        if (!mb[l]) continue;
        float4 kf = kvb[(size_t)l * 32];
        float s = q0 * kf.x + q1 * kf.y + q2 * kf.z + q3 * kf.w;
        float e;
        if (s > mx) {
            float c = __expf(mx - s);
            den *= c; a0 *= c; a1 *= c; a2 *= c; a3 *= c;
            mx = s; e = 1.f;
        } else e = __expf(s - mx);
        float4 vf = kvb[(size_t)l * 32 + 16];
        den += e; a0 += e * vf.x; a1 += e * vf.y; a2 += e * vf.z; a3 += e * vf.w;
    }
    float inv = 1.f / den;
    float* op = out + (size_t)nm * 64 + hh * 4;
    op[0] = a0 * inv; op[1] = a1 * inv; op[2] = a2 * inv; op[3] = a3 * inv;
}

// final: g -> leaky+LN (d1) -> leaky+LN (d2) -> heads.  One block per batch row.
__global__ void final_mlp(const float* __restrict__ gin,
                          const float* __restrict__ w1, const float* __restrict__ b1,
                          const float* __restrict__ g1, const float* __restrict__ bb1,
                          const float* __restrict__ w2, const float* __restrict__ b2,
                          const float* __restrict__ g2, const float* __restrict__ bb2,
                          const float* __restrict__ hw, const float* __restrict__ hb,
                          float* __restrict__ out)
{
    int b = blockIdx.x, tid = threadIdx.x;
    __shared__ float X[1024], Y[1024];
    __shared__ float r1[256], r2[256];
    for (int k = tid; k < 1024; k += 256) X[k] = gin[(size_t)b * 1024 + k];
    __syncthreads();

    float vals[4]; float ls, lq, mu, var, rs;

    ls = 0.f; lq = 0.f;
    for (int i = 0; i < 4; i++) {
        int f = tid + i * 256;
        float acc = b1[f];
        for (int k = 0; k < 1024; k++) acc += X[k] * w1[(size_t)k * 1024 + f];
        acc = leaky_(acc);
        vals[i] = acc; ls += acc; lq += acc * acc;
    }
    r1[tid] = ls; r2[tid] = lq; __syncthreads();
    for (int s = 128; s > 0; s >>= 1) {
        if (tid < s) { r1[tid] += r1[tid + s]; r2[tid] += r2[tid + s]; }
        __syncthreads();
    }
    mu = r1[0] / 1024.f; var = r2[0] / 1024.f - mu * mu; rs = rsqrtf(var + EPS_);
    for (int i = 0; i < 4; i++) {
        int f = tid + i * 256;
        Y[f] = (vals[i] - mu) * rs * g1[f] + bb1[f];
    }
    __syncthreads();

    ls = 0.f; lq = 0.f;
    for (int i = 0; i < 4; i++) {
        int f = tid + i * 256;
        float acc = b2[f];
        for (int k = 0; k < 1024; k++) acc += Y[k] * w2[(size_t)k * 1024 + f];
        acc = leaky_(acc);
        vals[i] = acc; ls += acc; lq += acc * acc;
    }
    r1[tid] = ls; r2[tid] = lq; __syncthreads();
    for (int s = 128; s > 0; s >>= 1) {
        if (tid < s) { r1[tid] += r1[tid + s]; r2[tid] += r2[tid + s]; }
        __syncthreads();
    }
    mu = r1[0] / 1024.f; var = r2[0] / 1024.f - mu * mu; rs = rsqrtf(var + EPS_);
    for (int i = 0; i < 4; i++) {
        int f = tid + i * 256;
        X[f] = (vals[i] - mu) * rs * g2[f] + bb2[f];
    }
    __syncthreads();

    for (int hh = 0; hh < 4; hh++) {
        float p = 0.f;
        for (int k = tid; k < 1024; k += 256) p += X[k] * hw[(size_t)hh * 1024 + k];
        __syncthreads();
        r1[tid] = p; __syncthreads();
        for (int s = 128; s > 0; s >>= 1) {
            if (tid < s) r1[tid] += r1[tid + s];
            __syncthreads();
        }
        if (tid == 0) out[b * 4 + hh] = r1[0] + hb[hh];
        __syncthreads();
    }
}

// ---------------------------------------------------------------------------
extern "C" void kernel_launch(void* const* d_in, const int* in_sizes, int n_in,
                              void* d_out, int out_size, void* d_ws, size_t ws_size,
                              hipStream_t stream)
{
    const float* x    = (const float*)d_in[0];
    const int*   ei   = (const int*)d_in[1];
    const float* efi  = (const float*)d_in[2];
    const float* prot = (const float*)d_in[4];
    const unsigned char* mask = (const unsigned char*)d_in[5];
    const int* srcp = ei;
    const int* dstp = ei + E_;

    const float* P[99];
    for (int i = 0; i < 99; i++) P[i] = (const float*)d_in[6 + i];
    // xe:0..3  ee:4..7  conv i:8+9i {wl,bl,wr,br,we,att,bias,ln_g,ln_b}
    // attn j:53+8j {wq,bq,wk,bk,wv,bv,wo,bo}  tail: 85.. pool/comb/d1/f1/d2/f2/heads
    const float* pool_w = P[85]; const float* pool_b = P[86];
    const float* comb_w = P[87]; const float* comb_b = P[88];

    // ---- workspace carve-out (f32 arena then f16 arena) ----
    float* w = (float*)d_ws;
    size_t o = 0;
    auto alloc = [&](size_t n) { float* p = w + o; o += n; return p; };
    float* h16  = alloc((size_t)NN_ * 32);     // padded to K=32
    float* h    = alloc((size_t)NN_ * 64);
    float* xl   = alloc((size_t)NN_ * 64);
    float* xr   = alloc((size_t)NN_ * 64);
    float* ef   = alloc((size_t)E_ * 2);
    float* cnt  = alloc(NN_);
    float* la   = alloc((size_t)NN_ * 2);
    float* den  = alloc(NN_);
    int*   mxi  = (int*)alloc(NN_);
    int*   s2   = (int*)alloc(ET_);
    int*   d2   = (int*)alloc(ET_);
    float* lgt  = alloc(ET_);
    float* av   = alloc(ET_);
    float* hout = alloc((size_t)NN_ * 64);
    float* qb   = alloc((size_t)NN_ * 64);
    float* kvb  = alloc((size_t)B_ * L_ * 128);
    float* ao   = alloc((size_t)NN_ * 64);
    float* bkv  = alloc((size_t)4 * 128);
    float* hdA  = alloc((size_t)NN_ * 1024);
    float* hdB  = alloc((size_t)NN_ * 1024);

    _Float16* hw16 = (_Float16*)(w + o);
    size_t ho = 0;
    auto halloc = [&](size_t n) { _Float16* p = hw16 + ho; ho += (n + 7) & ~(size_t)7; return p; };
    _Float16* t_wl[5]; _Float16* t_wr[5];
    for (int i = 0; i < 5; i++) { t_wl[i] = halloc(64 * 64); t_wr[i] = halloc(64 * 64); }
    _Float16* t_wq[4]; _Float16* t_wo[4]; _Float16* t_wkv[4];
    for (int j = 0; j < 4; j++) {
        t_wq[j] = halloc(64 * 64); t_wo[j] = halloc(64 * 64);
        t_wkv[j] = halloc((size_t)128 * PD_);
    }
    _Float16* t_pool = halloc((size_t)1024 * 64);
    _Float16* t_comb = halloc((size_t)1024 * 2048);

    // ---- one-time weight transpose/convert (f32 -> f16, N x Kpad) ----
    for (int i = 0; i < 5; i++) {
        const float** c = &P[8 + 9 * i];
        int cin = (i == 0) ? 16 : 64, cinp = (i == 0) ? 32 : 64;
        wt_prep<<<(64 * cinp + 255) / 256, 256, 0, stream>>>(c[0], t_wl[i], cin, 64, cinp);
        wt_prep<<<(64 * cinp + 255) / 256, 256, 0, stream>>>(c[2], t_wr[i], cin, 64, cinp);
    }
    for (int j = 0; j < 4; j++) {
        const float** a = &P[53 + 8 * j];
        wt_prep<<<(64 * 64 + 255) / 256, 256, 0, stream>>>(a[0], t_wq[j], 64, 64, 64);
        wt_prep<<<(64 * 64 + 255) / 256, 256, 0, stream>>>(a[6], t_wo[j], 64, 64, 64);
        kv_wt_prep<<<(128 * PD_ + 255) / 256, 256, 0, stream>>>(
            a[2], a[4], a[3], a[5], t_wkv[j], bkv + (size_t)j * 128);
    }
    wt_prep<<<(1024 * 64 + 255) / 256, 256, 0, stream>>>(pool_w, t_pool, 64, 1024, 64);
    wt_prep<<<(1024 * 2048 + 255) / 256, 256, 0, stream>>>(comb_w, t_comb, 2048, 1024, 2048);

    // ---- embeddings ----
    node_embed<<<B_, 256, 0, stream>>>(x, P[0], P[1], P[2], P[3], h16);
    edge_embed<<<E_ / 256, 256, 0, stream>>>(efi, P[4], P[5], P[6], P[7], ef);

    // ---- 5 GATv2 layers, cross-attention after layers 0..3 ----
    for (int i = 0; i < 5; i++) {
        const float** c = &P[8 + 9 * i];
        int Kc = (i == 0) ? 32 : 64;
        const float* hin = (i == 0) ? h16 : h;

        gemm_wmma<A_PLAIN, EPI_BIAS, 4><<<dim3(NN_ / 64, 1), 128, 0, stream>>>(
            hin, t_wl[i], c[1], nullptr, xl, NN_, Kc, 64);
        gemm_wmma<A_PLAIN, EPI_BIAS, 4><<<dim3(NN_ / 64, 1), 128, 0, stream>>>(
            hin, t_wr[i], c[3], nullptr, xr, NN_, Kc, 64);

        fill_u32<<<(NN_ + 255) / 256, 256, 0, stream>>>((unsigned*)cnt, 0u, NN_);
        fill_u32<<<(2 * NN_ + 255) / 256, 256, 0, stream>>>((unsigned*)la, 0u, 2 * NN_);
        fill_u32<<<(NN_ + 255) / 256, 256, 0, stream>>>((unsigned*)den, 0u, NN_);
        fill_u32<<<(NN_ + 255) / 256, 256, 0, stream>>>((unsigned*)mxi, 0x80000000u, NN_);
        fill_u32<<<(NN_ * 64 + 255) / 256, 256, 0, stream>>>((unsigned*)hout, 0u, NN_ * 64);

        gat_prep<<<E_ / 256, 256, 0, stream>>>(dstp, ef, cnt, la);
        gat_logits<<<(ET_ + 255) / 256, 256, 0, stream>>>(
            srcp, dstp, ef, cnt, la, xl, xr, c[4], c[5], lgt, s2, d2, mxi);
        gat_exp<<<(ET_ + 255) / 256, 256, 0, stream>>>(lgt, d2, mxi, av, den);
        gat_scatter<<<(ET_ * 64) / 256, 256, 0, stream>>>(av, den, s2, d2, xl, hout);
        gat_ln<<<B_, 256, 0, stream>>>(hout, c[6], c[7], c[8], h);

        if (i < 4) {
            const float** a = &P[53 + 8 * i];
            gemm_wmma<A_PLAIN, EPI_BIAS, 4><<<dim3(NN_ / 64, 1), 128, 0, stream>>>(
                h, t_wq[i], a[1], nullptr, qb, NN_, 64, 64);
            // fused K|V: single pass over the 31.5M-element protein layer
            gemm_wmma<A_PLAIN, EPI_BIAS, 8><<<dim3(B_ * L_ / 64, 1), 128, 0, stream>>>(
                prot + (size_t)i * B_ * L_ * PD_, t_wkv[i],
                bkv + (size_t)i * 128, nullptr, kvb, B_ * L_, PD_, 128);
            attn_softmax<<<(NN_ * H_) / 256, 256, 0, stream>>>(qb, kvb, mask, ao);
            gemm_wmma<A_PLAIN, EPI_RESID, 4><<<dim3(NN_ / 64, 1), 128, 0, stream>>>(
                ao, t_wo[i], a[7], h, h, NN_, 64, 64);
        }
    }

    // ---- pool (NT=8 -> grid.y = 8, halves L2 re-reads of h) ----
    gemm_wmma<A_PLAIN, EPI_BIAS, 8><<<dim3(NN_ / 64, 1024 / 128), 128, 0, stream>>>(
        h, t_pool, pool_b, nullptr, hdA, NN_, 64, 1024);

    // ---- 7-level symmetric tanh combine tree (pair-average fused in epilogue)
    float* cur = hdA; float* nxt = hdB;
    int S = M_;
    for (int t = 0; t < 7; t++) {
        int R = S >> 1;
        int Mr = 2 * B_ * R;   // both concat orders as GEMM rows; always %64==0
        gemm_wmma<A_COMBINE, EPI_TANHPAIR, 8><<<dim3(Mr / 64, 1024 / 128), 128, 0, stream>>>(
            cur, t_comb, comb_b, nullptr, nxt, Mr, 2048, 1024);
        float* tmp = cur; cur = nxt; nxt = tmp;
        S = R;
    }

    final_mlp<<<B_, 256, 0, stream>>>(cur, P[89], P[90], P[91], P[92],
                                      P[93], P[94], P[95], P[96],
                                      P[97], P[98], (float*)d_out);
}
